// LSTM_58153857188408
// MI455X (gfx1250) — matrix operand order repaired and checked by reading, not compiled, not run on hardware
//
#include <hip/hip_runtime.h>

// ---------------------------------------------------------------------------
// 3-layer LSTM forward for MI455X (gfx1250, wave32, WMMA).
//   B=32, T=512, H=512, L=3.  out[b][t][l*H+n] (f32).
// Per layer:
//   1) Wi/Wh f32 -> bf16 (layer-0 input converted once; later layers get
//      bf16 input written directly by the previous recurrence kernel)
//   2) xg[t][b][4H] = x_bf16 @ Wi^T + bi+bh   (WMMA bf16, 32x64 per wave)
//   3) recurrence split over 8 WGPs (8 blocks x 4 wave32): h ping-pongs in
//      L2-resident global bf16 buffers, staged into LDS each step via
//      global_load_async_to_lds_b128 (ASYNCcnt), device-scope atomic spin
//      barrier per timestep, c/h in registers in WMMA fragment layout.
// ---------------------------------------------------------------------------

typedef __attribute__((ext_vector_type(16))) __bf16 v16bf;
typedef __attribute__((ext_vector_type(8)))  __bf16 v8bf;
typedef __attribute__((ext_vector_type(8)))  float  v8f;

#define BB 32
#define TT 512
#define HH 512
#define GG 2048   // 4*H
#define LL 3
#define RBLK 8    // recurrence blocks (WGPs)
#define RTHR 128  // threads per recurrence block (4 wave32)

__device__ __forceinline__ unsigned short f2bf(float f) {
    unsigned u = __float_as_uint(f);
    u += 0x7fffu + ((u >> 16) & 1u);          // round-to-nearest-even
    return (unsigned short)(u >> 16);
}

__device__ __forceinline__ v16bf cat8(v8bf lo, v8bf hi) {
    return __builtin_shufflevector(lo, hi, 0,1,2,3,4,5,6,7,8,9,10,11,12,13,14,15);
}

__device__ __forceinline__ float sigm(float x)      { return 1.f / (1.f + __expf(-x)); }
__device__ __forceinline__ float tanh_fast(float x) { return 2.f / (1.f + __expf(-2.f * x)) - 1.f; }

// CDNA5 async global->LDS copy (ASYNCcnt-tracked), 16B per lane.
__device__ __forceinline__ void async_copy16(unsigned lds_off, const void* gptr) {
    asm volatile("global_load_async_to_lds_b128 %0, %1, off"
                 :: "v"(lds_off), "v"((unsigned long long)(size_t)gptr)
                 : "memory");
}
__device__ __forceinline__ void wait_async0() {
    asm volatile("s_wait_asynccnt 0x0" ::: "memory");
}

// --------------------------- f32 -> bf16 (strided rows) --------------------
__global__ void __launch_bounds__(256)
f32_to_bf16_rows(const float* __restrict__ src, unsigned short* __restrict__ dst,
                 int rows, int width, int srcStride)
{
    int i = blockIdx.x * 256 + threadIdx.x;
    if (i >= rows * width) return;
    int r = i / width;
    int c = i - r * width;
    dst[i] = f2bf(src[(size_t)r * srcStride + c]);
}

// --------------------------- per-layer recurrence state init ---------------
__global__ void __launch_bounds__(256)
init_state(unsigned short* __restrict__ h0, unsigned short* __restrict__ h1,
           unsigned* __restrict__ cnt)
{
    int i = blockIdx.x * 256 + threadIdx.x;
    if (i < BB * HH) { h0[i] = 0; h1[i] = 0; }
    if (i == 0) *cnt = 0u;
}

// --------------------------- xg GEMM: [B*T,512] x [2048,512]^T -------------
// Each wave: 32(M) x 64(N) strip, K=512; B fragments reused across 2 M-tiles
// (8 WMMA per 12 b128 loads). Output xg[t][b][4H] so the recurrence reads a
// contiguous 32x2048 slab per step.
__global__ void __launch_bounds__(256)
gemm_xg(const unsigned short* __restrict__ xb,   // [B*T][H] bf16
        const unsigned short* __restrict__ wb,   // [4H][H]  bf16 (Wi row-major)
        const float* __restrict__ bi, const float* __restrict__ bh,
        float* __restrict__ xg)                  // [T][B][4H] f32
{
    const int lane = threadIdx.x & 31;
    const int wid  = threadIdx.x >> 5;
    const int gw   = blockIdx.x * 8 + wid;       // global wave id (16384)
    const int m0   = (gw >> 5) << 5;             // 512 M-strips of 32
    const int n0   = (gw & 31) << 6;             // 32 N-strips of 64
    const int l16  = lane & 15;
    const int sel  = lane >> 4;

    v8f acc[2][4] = {};
    const unsigned short* arow0 = xb + (size_t)(m0 + l16) * HH;
    const unsigned short* arow1 = arow0 + (size_t)16 * HH;

    for (int k0 = 0; k0 < HH; k0 += 32) {
        // A frags: lane m=l16, contiguous K {k0+sel*16..+15}
        v16bf a0 = cat8(*(const v8bf*)(arow0 + k0 + sel * 16),
                        *(const v8bf*)(arow0 + k0 + 8 + sel * 16));
        v16bf a1 = cat8(*(const v8bf*)(arow1 + k0 + sel * 16),
                        *(const v8bf*)(arow1 + k0 + 8 + sel * 16));
        __builtin_prefetch(arow0 + k0 + 64, 0, 3);      // global_prefetch
#pragma unroll
        for (int nt = 0; nt < 4; ++nt) {
            // B frag: lane = column n, contiguous 16 K values per half-wave
            const unsigned short* bp =
                wb + (size_t)(n0 + nt * 16 + l16) * HH + k0 + sel * 16;
            v16bf b = *(const v16bf*)bp;
            acc[0][nt] = __builtin_amdgcn_wmma_f32_16x16x32_bf16(
                false, a0, false, b, (short)0, acc[0][nt], false, false);
            acc[1][nt] = __builtin_amdgcn_wmma_f32_16x16x32_bf16(
                false, a1, false, b, (short)0, acc[1][nt], false, false);
        }
    }

#pragma unroll
    for (int nt = 0; nt < 4; ++nt) {
        const int n = n0 + nt * 16 + l16;
        const float bias = bi[n] + bh[n];
#pragma unroll
        for (int mt = 0; mt < 2; ++mt)
#pragma unroll
            for (int r = 0; r < 8; ++r) {
                const int M = m0 + mt * 16 + r + sel * 8;  // row = b*T + t
                const int b = M >> 9;                      // /T
                const int t = M & (TT - 1);
                xg[((size_t)t * BB + b) * GG + n] = acc[mt][nt][r] + bias;
            }
    }
}

// --------------------------- multi-WGP recurrence --------------------------
// 8 blocks x 128 threads (4 wave32 each -> 1 wave per SIMD on 8 WGPs).
// Global wave w owns hidden columns [16w,16w+16) and all 4 gate tiles for
// batch rows 0..31. Per step: async-stage h (32KB bf16) into LDS once per
// block, 128 v_wmma per wave, pointwise LSTM in registers, write h to the
// ping-pong buffer, then device-scope spin barrier.
__global__ void __launch_bounds__(RTHR)
lstm_steps(const float* __restrict__ xg,           // [T][B][4H] (incl. biases)
           const unsigned short* __restrict__ whb, // [4H][H] bf16 (Wh)
           const int* __restrict__ lengths,        // [B]
           float* __restrict__ out,                // d_out + l*H, stride L*H
           unsigned short* __restrict__ h0,        // ping
           unsigned short* __restrict__ h1,        // pong
           unsigned* __restrict__ barcnt,          // spin-barrier counter
           unsigned short* __restrict__ xbnext)    // bf16 input for next layer
{
    __shared__ unsigned short hlds[BB * HH];       // 32KB staged h
    const int tid  = threadIdx.x;
    const int lane = tid & 31;
    const int gwid = blockIdx.x * (RTHR / 32) + (tid >> 5);  // 0..31
    const int l16  = lane & 15;
    const int sel  = lane >> 4;
    const int n0   = gwid * 16;
    const int n    = n0 + l16;

    int   len[2][8];
    float cst[2][8];
    float hst[2][8];
#pragma unroll
    for (int mt = 0; mt < 2; ++mt)
#pragma unroll
        for (int r = 0; r < 8; ++r) {
            len[mt][r] = lengths[mt * 16 + r + sel * 8];
            cst[mt][r] = 0.f;
            hst[mt][r] = 0.f;
        }

    for (int t = 0; t < TT; ++t) {
        const unsigned short* hsrc = (t & 1) ? h1 : h0;
        unsigned short*       hdst = (t & 1) ? h0 : h1;

        // stage full h into LDS (async, ASYNCcnt-tracked): 32KB / block
        {
            const char* gsrc = (const char*)hsrc;
            char* lbase = (char*)hlds;
            for (int off = tid * 16; off < BB * HH * 2; off += RTHR * 16)
                async_copy16((unsigned)(size_t)(lbase + off), gsrc + off);
            wait_async0();
            __syncthreads();
        }

        v8f acc[2][4] = {};
        for (int k0 = 0; k0 < HH; k0 += 32) {
            const unsigned short* ap0 = &hlds[(unsigned)(l16)      * HH + k0 + sel * 16];
            const unsigned short* ap1 = &hlds[(unsigned)(16 + l16) * HH + k0 + sel * 16];
            v16bf a0 = cat8(*(const v8bf*)ap0, *(const v8bf*)(ap0 + 8));
            v16bf a1 = cat8(*(const v8bf*)ap1, *(const v8bf*)(ap1 + 8));
#pragma unroll
            for (int g = 0; g < 4; ++g) {
                const unsigned short* bp =
                    whb + (size_t)(g * HH + n) * HH + k0 + sel * 16;
                v16bf b = *(const v16bf*)bp;     // L2-resident Wh
                acc[0][g] = __builtin_amdgcn_wmma_f32_16x16x32_bf16(
                    false, a0, false, b, (short)0, acc[0][g], false, false);
                acc[1][g] = __builtin_amdgcn_wmma_f32_16x16x32_bf16(
                    false, a1, false, b, (short)0, acc[1][g], false, false);
            }
        }

        const float* xgt = xg + (size_t)t * BB * GG;   // contiguous 32x2048
#pragma unroll
        for (int mt = 0; mt < 2; ++mt) {
#pragma unroll
            for (int r = 0; r < 8; ++r) {
                const int M = mt * 16 + r + sel * 8;   // batch row
                const float* row = xgt + (size_t)M * GG + n;
                float iv = sigm(acc[mt][0][r] + row[0]);
                float fv = sigm(acc[mt][1][r] + row[HH]);
                float gv = tanh_fast(acc[mt][2][r] + row[2 * HH]);
                float ov = sigm(acc[mt][3][r] + row[3 * HH]);
                float cy = fv * cst[mt][r] + iv * gv;
                float hy = ov * tanh_fast(cy);
                const bool msk = t < len[mt][r];
                cst[mt][r] = msk ? cy : cst[mt][r];
                hst[mt][r] = msk ? hy : hst[mt][r];
                const float ov_out = msk ? hy : 0.f;
                out[((size_t)M * TT + t) * (LL * HH) + n] = ov_out;
                // bf16 input for the next layer (fused conversion)
                xbnext[((size_t)M * TT + t) * HH + n] = f2bf(ov_out);
                // next-step h into the ping-pong buffer
                hdst[(unsigned)M * HH + n] = f2bf(hst[mt][r]);
            }
        }

        // device-scope barrier: all 8 blocks finished step t
        __threadfence();
        __syncthreads();
        if (tid == 0) {
            __hip_atomic_fetch_add(barcnt, 1u, __ATOMIC_RELEASE,
                                   __HIP_MEMORY_SCOPE_AGENT);
            const unsigned target = (unsigned)RBLK * (unsigned)(t + 1);
            while (__hip_atomic_load(barcnt, __ATOMIC_ACQUIRE,
                                     __HIP_MEMORY_SCOPE_AGENT) < target)
                __builtin_amdgcn_s_sleep(1);
        }
        __syncthreads();
    }
}

// --------------------------- host launcher ---------------------------------
extern "C" void kernel_launch(void* const* d_in, const int* in_sizes, int n_in,
                              void* d_out, int out_size, void* d_ws, size_t ws_size,
                              hipStream_t stream) {
    const float* x       = (const float*)d_in[0];
    const float* Wi      = (const float*)d_in[1];
    const float* bi      = (const float*)d_in[2];
    const float* Wh      = (const float*)d_in[3];
    const float* bh      = (const float*)d_in[4];
    const int*   lengths = (const int*)d_in[5];
    float* out = (float*)d_out;
    char*  ws  = (char*)d_ws;

    size_t off = 0;
    auto take = [&](size_t bytes) {
        size_t o = off; off += (bytes + 255) & ~(size_t)255; return o;
    };
    unsigned short* xb  = (unsigned short*)(ws + take((size_t)BB * TT * HH * 2));
    unsigned short* wib = (unsigned short*)(ws + take((size_t)GG * HH * 2));
    unsigned short* whb = (unsigned short*)(ws + take((size_t)GG * HH * 2));
    unsigned short* h0  = (unsigned short*)(ws + take((size_t)BB * HH * 2));
    unsigned short* h1  = (unsigned short*)(ws + take((size_t)BB * HH * 2));
    unsigned*       cnt = (unsigned*)(ws + take(256));
    float*          xg  = (float*)(ws + take((size_t)TT * BB * GG * 4));
    (void)ws_size; // assumes ws_size >= ~156MB

    const int rowsX = BB * TT;                        // 16384
    // layer-0 input -> bf16 (later layers written by lstm_steps directly)
    f32_to_bf16_rows<<<(rowsX * HH + 255) / 256, 256, 0, stream>>>(
        x, xb, rowsX, HH, HH);

    for (int l = 0; l < LL; ++l) {
        f32_to_bf16_rows<<<(GG * HH + 255) / 256, 256, 0, stream>>>(
            Wi + (size_t)l * GG * HH, wib, GG, HH, HH);
        f32_to_bf16_rows<<<(GG * HH + 255) / 256, 256, 0, stream>>>(
            Wh + (size_t)l * GG * HH, whb, GG, HH, HH);
        // xg = x @ Wi^T + bi + bh : 512 M-strips * 32 N-strips / 8 waves
        gemm_xg<<<(rowsX / 32) * (GG / 64) / 8, 256, 0, stream>>>(
            xb, wib, bi + (size_t)l * GG, bh + (size_t)l * GG, xg);
        // recurrence (reads xb fully consumed by gemm; overwrites xb for l+1)
        init_state<<<64, 256, 0, stream>>>(h0, h1, cnt);
        lstm_steps<<<RBLK, RTHR, 0, stream>>>(
            xg, whb, lengths, out + (size_t)l * HH, h0, h1, cnt, xb);
    }
}
